// TransitionGNN_18330920419718
// MI455X (gfx1250) — compile-verified, matrix operand order checked by reading, not run on hardware
//
#include <hip/hip_runtime.h>
#include <hip/hip_bf16.h>

// MI455X / gfx1250, wave32. Matrix work via v_wmma_f32_16x16x32_bf16
// (bf16 inputs, f32 accumulate) — compute-bound GEMMs, bandwidth is irrelevant.
//
// Weights are pre-swizzled into WMMA-fragment-native order so every B fragment
// is two contiguous global_load_b128 per lane (fully coalesced per wave).

typedef __attribute__((ext_vector_type(16))) __bf16       v16bf;
typedef __attribute__((ext_vector_type(8)))  float        v8f;
typedef __attribute__((ext_vector_type(8)))  unsigned int v8u;

#define LN_EPS 1e-5f

__device__ __forceinline__ v16bf as_v16bf(v8u u) {
  union { v8u u; v16bf b; } c; c.u = u; return c.b;
}

// round-to-nearest-even f32 -> bf16, packed pair into one dword
__device__ __forceinline__ unsigned int pack_bf16(float a, float b) {
  unsigned int ua = __float_as_uint(a);
  unsigned int ub = __float_as_uint(b);
  unsigned int ra = (ua + 0x7FFFu + ((ua >> 16) & 1u)) >> 16;
  unsigned int rb = (ub + 0x7FFFu + ((ub >> 16) & 1u)) >> 16;
  return (ra & 0xFFFFu) | (rb << 16);
}

// One GEMM stage: H[16][N] = A[16][Kpad] @ W[Kpad][N] + bias.
//   Ap : LDS, bf16 pairs packed along K, row stride Kpad/2 dwords.
//   Wf : global, fragment-swizzled bf16 weights:
//        dword index = ((kt*N + n)*2 + hi)*8 + v   (kt = K-chunk of 32)
// 8 waves; wave w owns N-tiles [w*NT, w*NT+NT).
// A-fragment layout per ISA (16-bit A 16x32): lane holds row M=lane&15;
// lane[4] selects K 0-7/16-23 vs 8-15/24-31. B symmetric (lane = column N).
template <int NT>
__device__ __forceinline__ void gemm_tiles(
    const unsigned int* __restrict__ Wf,
    const float* __restrict__ bias,
    const unsigned int* Ap,
    float* H,
    int Kpad, int N, int tid)
{
  const int wave = tid >> 5;
  const int lane = tid & 31;
  const int lo   = lane & 15;
  const int hi   = lane >> 4;
  const int ld2  = Kpad >> 1;
  const uint4* __restrict__ W4 = (const uint4*)Wf;

  v8f acc[NT];
#pragma unroll
  for (int t = 0; t < NT; ++t) {
    const int n = (wave * NT + t) * 16 + lo;
    const float bv = bias[n];
#pragma unroll
    for (int v = 0; v < 8; ++v) acc[t][v] = bv;
  }

  const int nChunks = Kpad >> 5;
  for (int kt = 0; kt < nChunks; ++kt) {
    v8u au;
#pragma unroll
    for (int v = 0; v < 8; ++v) {
      const int off2 = ((v < 4) ? v : v + 4) + hi * 4;  // packed-pair index in chunk
      au[v] = Ap[lo * ld2 + kt * 16 + off2];
    }
    const v16bf af = as_v16bf(au);
#pragma unroll
    for (int t = 0; t < NT; ++t) {
      const int n = (wave * NT + t) * 16 + lo;
      const size_t fi = (((size_t)kt * N + n) * 2 + hi) * 2;  // uint4 units
      const uint4 b0 = W4[fi];
      const uint4 b1 = W4[fi + 1];
      v8u bu;
      bu[0] = b0.x; bu[1] = b0.y; bu[2] = b0.z; bu[3] = b0.w;
      bu[4] = b1.x; bu[5] = b1.y; bu[6] = b1.z; bu[7] = b1.w;
      acc[t] = __builtin_amdgcn_wmma_f32_16x16x32_bf16(
          false, af, false, as_v16bf(bu), (short)0, acc[t], false, false);
    }
  }

  // C/D layout: VGPR v, lanes 0-15 -> row v, lanes 16-31 -> row v+8; col = lane&15
#pragma unroll
  for (int t = 0; t < NT; ++t) {
    const int n = (wave * NT + t) * 16 + lo;
#pragma unroll
    for (int v = 0; v < 8; ++v) H[(v + hi * 8) * N + n] = acc[t][v];
  }
  __syncthreads();
}

// relu(H[16][N]) -> Ap bf16 pairs (row stride N/2 dwords)
__device__ __forceinline__ void repack_relu(const float* H, unsigned int* Ap, int N, int tid) {
  const int half = N >> 1;
  for (int idx = tid; idx < 16 * half; idx += 256) {
    const int m  = idx / half;
    const int k2 = idx - m * half;
    float a = H[m * N + 2 * k2];
    float b = H[m * N + 2 * k2 + 1];
    a = a > 0.f ? a : 0.f;
    b = b > 0.f ? b : 0.f;
    Ap[idx] = pack_bf16(a, b);
  }
  __syncthreads();
}

// per-row LayerNorm + relu, in place. Wave w handles rows 2w, 2w+1 (wave32).
__device__ __forceinline__ void ln_relu(float* H, const float* __restrict__ g,
                                        const float* __restrict__ beta, int N, int tid) {
  const int wave = tid >> 5;
  const int lane = tid & 31;
#pragma unroll
  for (int rr = 0; rr < 2; ++rr) {
    const int r = wave * 2 + rr;
    float s = 0.f, s2 = 0.f;
    for (int n = lane; n < N; n += 32) {
      const float z = H[r * N + n];
      s += z; s2 += z * z;
    }
#pragma unroll
    for (int o = 16; o > 0; o >>= 1) {
      s  += __shfl_xor(s, o, 32);
      s2 += __shfl_xor(s2, o, 32);
    }
    const float mu   = s / (float)N;
    const float var  = s2 / (float)N - mu * mu;
    const float rstd = rsqrtf(var + LN_EPS);
    for (int n = lane; n < N; n += 32) {
      const float z = H[r * N + n];
      float y = (z - mu) * rstd * g[n] + beta[n];
      H[r * N + n] = y > 0.f ? y : 0.f;
    }
  }
  __syncthreads();
}

// f32 weights [K][N] -> fragment-swizzled packed bf16:
//   out dword index ((kt*N + n)*2 + hi)*8 + v  holds K-pair
//   k2 = kt*16 + ((v<4)?v:v+4) + hi*4  of column n  (zero-padded to Kpad).
__global__ void pack_weights_frag_kernel(const float* __restrict__ W,
                                         unsigned int* __restrict__ Wf,
                                         int K, int Kpad, int N) {
  const int total = (Kpad >> 1) * N;
  for (int idx = blockIdx.x * blockDim.x + threadIdx.x; idx < total;
       idx += gridDim.x * blockDim.x) {
    const int v  = idx & 7;
    const int hi = (idx >> 3) & 1;
    const int t  = idx >> 4;
    const int n  = t % N;
    const int kt = t / N;
    const int off2 = ((v < 4) ? v : v + 4) + hi * 4;
    const int k = (kt * 16 + off2) * 2;
    const float f0 = (k < K)     ? W[(size_t)k * N + n]       : 0.f;
    const float f1 = (k + 1 < K) ? W[(size_t)(k + 1) * N + n] : 0.f;
    Wf[idx] = pack_bf16(f0, f1);
  }
}

// One block per source node: its 15 edges as a padded M=16 tile through the
// 3-layer edge MLP, then reduce rows 0..14 -> agg[src]. Avoids materializing
// the 252 MB edge_attr tensor entirely.
__global__ __launch_bounds__(256) void edge_mlp_agg_kernel(
    const float* __restrict__ states,
    const unsigned int* __restrict__ eW1f, const float* __restrict__ eb1,
    const unsigned int* __restrict__ eW2f, const float* __restrict__ eb2,
    const float* __restrict__ eg,  const float* __restrict__ ebeta,
    const unsigned int* __restrict__ eW3f, const float* __restrict__ eb3,
    float* __restrict__ agg)
{
  __shared__ unsigned int Ap[16 * 256];   // activations, bf16 pairs (max K=512)
  __shared__ float        H[16 * 512];    // f32 layer output
  const int tid = threadIdx.x;
  const int s = blockIdx.x;               // source node, 0..8191
  const int b = s >> 4;
  const int i = s & 15;

  // Edge m (m=0..14): input = [states[src] , states[dst_m]], K=256. Row 15 = 0.
  for (int idx = tid; idx < 16 * 128; idx += 256) {
    const int m  = idx >> 7;
    const int k2 = idx & 127;
    const int k  = k2 * 2;
    float f0 = 0.f, f1 = 0.f;
    if (m < 15) {
      if (k < 128) {
        f0 = states[(size_t)s * 128 + k];
        f1 = states[(size_t)s * 128 + k + 1];
      } else {
        const int j = m + (m >= i ? 1 : 0);       // skip self-loop
        const size_t c = (size_t)(b * 16 + j) * 128 + (k - 128);
        f0 = states[c];
        f1 = states[c + 1];
      }
    }
    Ap[idx] = pack_bf16(f0, f1);
  }
  __syncthreads();

  gemm_tiles<4>(eW1f, eb1, Ap, H, 256, 512, tid);
  repack_relu(H, Ap, 512, tid);
  gemm_tiles<4>(eW2f, eb2, Ap, H, 512, 512, tid);
  ln_relu(H, eg, ebeta, 512, tid);
  repack_relu(H, Ap, 512, tid);
  gemm_tiles<4>(eW3f, eb3, Ap, H, 512, 512, tid);

  for (int n = tid; n < 512; n += 256) {
    float sum = 0.f;
#pragma unroll
    for (int m = 0; m < 15; ++m) sum += H[m * 512 + n];
    agg[(size_t)s * 512 + n] = sum;
  }
}

// One block per 16 nodes: [obs(128) | one-hot act(4) | agg(512) | pad(28)] -> MLP.
__global__ __launch_bounds__(256) void node_mlp_kernel(
    const float* __restrict__ states,
    const int* __restrict__ action,
    const float* __restrict__ agg,
    const unsigned int* __restrict__ nW1f, const float* __restrict__ nb1,
    const unsigned int* __restrict__ nW2f, const float* __restrict__ nb2,
    const float* __restrict__ ng,  const float* __restrict__ nbeta,
    const unsigned int* __restrict__ nW3f, const float* __restrict__ nb3,
    float* __restrict__ out)
{
  __shared__ unsigned int Ap[16 * 336];   // K=672 packed
  __shared__ float        H[16 * 512];
  const int tid = threadIdx.x;
  const int g0 = blockIdx.x * 16;

  for (int idx = tid; idx < 16 * 336; idx += 256) {
    const int m  = idx / 336;
    const int k2 = idx - m * 336;
    const int k  = 2 * k2;                 // region boundaries 128/132/644 all even
    const int gn = g0 + m;
    float f0 = 0.f, f1 = 0.f;
    if (k < 128) {
      f0 = states[(size_t)gn * 128 + k];
      f1 = states[(size_t)gn * 128 + k + 1];
    } else if (k < 132) {
      const int av = action[gn >> 4];
      const int base = (gn & 15) * 4;      // one_hot(action,64).reshape(B*O, 4)
      f0 = (av == base + (k - 128)) ? 1.f : 0.f;
      f1 = (av == base + (k - 127)) ? 1.f : 0.f;
    } else if (k < 644) {
      f0 = agg[(size_t)gn * 512 + (k - 132)];
      f1 = agg[(size_t)gn * 512 + (k - 131)];
    }
    Ap[idx] = pack_bf16(f0, f1);
  }
  __syncthreads();

  gemm_tiles<4>(nW1f, nb1, Ap, H, 672, 512, tid);
  repack_relu(H, Ap, 512, tid);
  gemm_tiles<4>(nW2f, nb2, Ap, H, 512, 512, tid);
  ln_relu(H, ng, nbeta, 512, tid);
  repack_relu(H, Ap, 512, tid);
  gemm_tiles<1>(nW3f, nb3, Ap, H, 512, 128, tid);   // N=128: 1 tile per wave

  for (int idx = tid; idx < 16 * 128; idx += 256) {
    const int m = idx >> 7;
    const int n = idx & 127;
    out[(size_t)(g0 + m) * 128 + n] = H[m * 128 + n];
  }
}

extern "C" void kernel_launch(void* const* d_in, const int* in_sizes, int n_in,
                              void* d_out, int out_size, void* d_ws, size_t ws_size,
                              hipStream_t stream) {
  const float* states = (const float*)d_in[0];
  const int*   action = (const int*)d_in[1];
  const float* eW1 = (const float*)d_in[2];
  const float* eb1 = (const float*)d_in[3];
  const float* eW2 = (const float*)d_in[4];
  const float* eb2 = (const float*)d_in[5];
  const float* eg  = (const float*)d_in[6];
  const float* ebt = (const float*)d_in[7];
  const float* eW3 = (const float*)d_in[8];
  const float* eb3 = (const float*)d_in[9];
  const float* nW1 = (const float*)d_in[10];
  const float* nb1 = (const float*)d_in[11];
  const float* nW2 = (const float*)d_in[12];
  const float* nb2 = (const float*)d_in[13];
  const float* ng  = (const float*)d_in[14];
  const float* nbt = (const float*)d_in[15];
  const float* nW3 = (const float*)d_in[16];
  const float* nb3 = (const float*)d_in[17];

  char* ws = (char*)d_ws;
  float* agg = (float*)ws;                                   // 8192*512 f32 = 16 MiB
  size_t off = (size_t)8192 * 512 * 4;
  unsigned int* eW1f = (unsigned int*)(ws + off); off += (size_t)128 * 512 * 4;
  unsigned int* eW2f = (unsigned int*)(ws + off); off += (size_t)256 * 512 * 4;
  unsigned int* eW3f = (unsigned int*)(ws + off); off += (size_t)256 * 512 * 4;
  unsigned int* nW1f = (unsigned int*)(ws + off); off += (size_t)336 * 512 * 4;
  unsigned int* nW2f = (unsigned int*)(ws + off); off += (size_t)256 * 512 * 4;
  unsigned int* nW3f = (unsigned int*)(ws + off); off += (size_t)256 * 128 * 4;
  (void)ws_size; (void)in_sizes; (void)n_in; (void)out_size;

  pack_weights_frag_kernel<<<256, 256, 0, stream>>>(eW1, eW1f, 256, 256, 512);
  pack_weights_frag_kernel<<<256, 256, 0, stream>>>(eW2, eW2f, 512, 512, 512);
  pack_weights_frag_kernel<<<256, 256, 0, stream>>>(eW3, eW3f, 512, 512, 512);
  pack_weights_frag_kernel<<<512, 256, 0, stream>>>(nW1, nW1f, 644, 672, 512);
  pack_weights_frag_kernel<<<256, 256, 0, stream>>>(nW2, nW2f, 512, 512, 512);
  pack_weights_frag_kernel<<<128, 256, 0, stream>>>(nW3, nW3f, 512, 512, 128);

  edge_mlp_agg_kernel<<<8192, 256, 0, stream>>>(states, eW1f, eb1, eW2f, eb2,
                                                eg, ebt, eW3f, eb3, agg);
  node_mlp_kernel<<<512, 256, 0, stream>>>(states, action, agg, nW1f, nb1,
                                           nW2f, nb2, ng, nbt, nW3f, nb3,
                                           (float*)d_out);
}
